// Non_Local_13735305413163
// MI455X (gfx1250) — matrix operand order repaired and checked by reading, not compiled, not run on hardware
//
#include <hip/hip_runtime.h>
#include <hip/hip_fp16.h>

typedef __attribute__((ext_vector_type(16))) _Float16 v16h;
typedef __attribute__((ext_vector_type(8)))  _Float16 v8h;
typedef __attribute__((ext_vector_type(8)))  float    v8f;
typedef __attribute__((ext_vector_type(4)))  unsigned int v4u;
typedef __attribute__((ext_vector_type(8)))  int      v8i;
typedef __attribute__((ext_vector_type(4)))  int      v4i;

#define N_PIX 4096
#define C_CH  256
#define D_CH  64

// padded LDS row strides (elements) chosen via TDM pad feature:
//   K rows: 64 f16 = 32 dwords + 4 dwords pad -> 36 dwords = 72 elem (144B, 16B-aligned)
//   V rows: 32 f16 = 16 dwords + 4 dwords pad -> 20 dwords = 40 elem (80B, 16B-aligned)
#define KROW 72
#define VROW 40

static __device__ __forceinline__ v8f wmma16(v16h a, v16h b, v8f c) {
  // D = A(16x32 f16) * B(32x16 f16) + C(16x16 f32)
  return __builtin_amdgcn_wmma_f32_16x16x32_f16(false, a, false, b, (short)0, c,
                                                false, false);
}

static __device__ __forceinline__ v16h cat8(v8h lo, v8h hi) {
  return __builtin_shufflevector(lo, hi, 0, 1, 2, 3, 4, 5, 6, 7, 8, 9, 10, 11,
                                 12, 13, 14, 15);
}

// ---------------------------------------------------------------------------
// Tensor Data Mover: 2-D tile (f16) global -> LDS, with per-row LDS padding.
// D# layout per CDNA5 ISA 8.3/8.4:
//   g0: [1:0]=count=1, [63:32]=lds_addr, [120:64]=global_addr, [127:126]=type2
//   g1: [17:16]=data_size(1=2B), [20]=pad_en, [24:22]=pad_interval,
//       [31:25]=pad_amount, [79:48]=tensor_dim0, [111:80]=tensor_dim1,
//       [127:112]=tile_dim0, [143:128]=tile_dim1, [207:160]=dim0_stride
// ---------------------------------------------------------------------------
static __device__ __forceinline__ void tdm_load_tile_2d(
    unsigned lds_byte_off, const _Float16* gptr, unsigned tensor_d0,
    unsigned tensor_d1, unsigned tile_d0, unsigned tile_d1,
    unsigned row_stride_elems, unsigned pad_interval_code,
    unsigned pad_amount_code) {
  unsigned long long ga = (unsigned long long)(uintptr_t)gptr;
  v4u g0;
  g0[0] = 1u;                                   // count=1, user descriptor
  g0[1] = lds_byte_off;                         // LDS byte address
  g0[2] = (unsigned)ga;                         // global addr [31:0]
  g0[3] = (unsigned)((ga >> 32) & 0x01FFFFFFu) | (2u << 30);  // [56:32]|type=2
  v8i g1;
  g1[0] = (int)((1u << 16) | (1u << 20) | (pad_interval_code << 22) |
                (pad_amount_code << 25));       // 2B elems, pad enabled
  g1[1] = (int)((tensor_d0 & 0xFFFFu) << 16);
  g1[2] = (int)(((tensor_d0 >> 16) & 0xFFFFu) | ((tensor_d1 & 0xFFFFu) << 16));
  g1[3] = (int)(((tensor_d1 >> 16) & 0xFFFFu) | (tile_d0 << 16));
  g1[4] = (int)tile_d1;                         // tile_dim1 (tile_dim2 = 0)
  g1[5] = (int)row_stride_elems;                // tensor_dim0_stride low 32
  g1[6] = 0;
  g1[7] = 0;
  v4i gz = {0, 0, 0, 0};
#if defined(__clang_major__) && __clang_major__ >= 23
  v8i gz8 = {0, 0, 0, 0, 0, 0, 0, 0};
  __builtin_amdgcn_tensor_load_to_lds(g0, g1, gz, gz, gz8, 0);
#else
  __builtin_amdgcn_tensor_load_to_lds(g0, g1, gz, gz, 0);
#endif
}

// ---------------------------------------------------------------------------
// Phase 0: convert weights to f16 once (wq/wk/wv [64x256], wsa [256x64])
// ---------------------------------------------------------------------------
__global__ __launch_bounds__(256) void prep_weights(
    const float* __restrict__ wq, const float* __restrict__ wk,
    const float* __restrict__ wv, const float* __restrict__ wsa,
    _Float16* __restrict__ wqh, _Float16* __restrict__ wkh,
    _Float16* __restrict__ wvh, _Float16* __restrict__ wsah) {
  int i = blockIdx.x * 256 + threadIdx.x;
  if (i < 16384) {
    wqh[i]  = (_Float16)wq[i];
    wkh[i]  = (_Float16)wk[i];
    wvh[i]  = (_Float16)wv[i];
    wsah[i] = (_Float16)wsa[i];
  }
}

// ---------------------------------------------------------------------------
// Phase 1: QKV projections via WMMA.
//   Qt, Kt stored as [N,64] f16 (pixel-major), V stored as [64,N] f16.
//   x tile [256c x 32n] staged transposed in LDS (conflict-free pad 258).
// ---------------------------------------------------------------------------
__global__ __launch_bounds__(256) void qkv_kernel(
    const float* __restrict__ x, const _Float16* __restrict__ wqh,
    const _Float16* __restrict__ wkh, const _Float16* __restrict__ wvh,
    const float* __restrict__ bq, const float* __restrict__ bk,
    const float* __restrict__ bv, _Float16* __restrict__ Qt,
    _Float16* __restrict__ Kt, _Float16* __restrict__ V) {
  __shared__ float Xs[32 * 258];
  const int b  = blockIdx.y;
  const int n0 = blockIdx.x << 5;
  for (int e = threadIdx.x; e < 256 * 32; e += 256) {
    int c = e >> 5, j = e & 31;
    Xs[j * 258 + c] = x[(((size_t)b * 256 + c) << 12) + n0 + j];
  }
  __syncthreads();
  const int wave = threadIdx.x >> 5;
  const int lane = threadIdx.x & 31;
  const int m    = lane & 15;
  const int h    = lane >> 4;
#pragma unroll
  for (int q = 0; q < 3; ++q) {
    int job = wave * 3 + q;          // 24 jobs: 3 mats x 4 d-tiles x 2 n-tiles
    int mat = job >> 3;
    int dt  = (job >> 1) & 3;
    int nt  = job & 1;
    int d0  = dt << 4;
    int nb  = nt << 4;
    const _Float16* W = (mat == 0) ? wqh : (mat == 1) ? wkh : wvh;
    v8f acc = {};
    if (mat < 2) {
      // Qt/Kt: M=n, N=d, K=c.  A = x^T rows from LDS, B = W rows.
#pragma unroll
      for (int c0 = 0; c0 < 256; c0 += 32) {
        const float* row = &Xs[(nb + m) * 258 + c0];
        v16h a;
#pragma unroll
        for (int e2 = 0; e2 < 8; ++e2) {
          a[e2]     = (_Float16)row[h * 8 + e2];
          a[8 + e2] = (_Float16)row[16 + h * 8 + e2];
        }
        v16h bf = cat8(*(const v8h*)(W + (d0 + m) * 256 + c0 + h * 8), *(const v8h*)(W + (d0 + m) * 256 + c0 + 16 + h * 8));
        acc = wmma16(a, bf, acc);
      }
      const float bb = ((mat == 0) ? bq : bk)[d0 + m];
      _Float16* dst = ((mat == 0) ? Qt : Kt) + (size_t)b * N_PIX * D_CH;
#pragma unroll
      for (int r = 0; r < 8; ++r)
        dst[(size_t)(n0 + nb + r + 8 * h) * 64 + d0 + m] =
            (_Float16)(acc[r] + bb);
    } else {
      // V: M=d, N=n, K=c.  A = W rows, B = x^T rows from LDS.
#pragma unroll
      for (int c0 = 0; c0 < 256; c0 += 32) {
        const _Float16* wrow = W + (d0 + m) * 256 + c0;
        v16h a = cat8(*(const v8h*)(wrow + h * 16),
                      *(const v8h*)(wrow + h * 16 + 8));
        const float* xrow = &Xs[(nb + m) * 258 + c0 + h * 16];
        v16h bf;
#pragma unroll
        for (int e2 = 0; e2 < 16; ++e2) bf[e2] = (_Float16)xrow[e2];
        acc = wmma16(a, bf, acc);
      }
      _Float16* dst = V + (size_t)b * N_PIX * D_CH;
#pragma unroll
      for (int r = 0; r < 8; ++r) {
        int d = d0 + r + 8 * h;
        dst[(size_t)d * N_PIX + n0 + nb + m] = (_Float16)(acc[r] + bv[d]);
      }
    }
  }
}

// ---------------------------------------------------------------------------
// Phase 2: fused flash attention.  8 waves/block, one 16-query tile per wave,
// all waves stream the SAME j-blocks.  K (32x64) and V (64x32) tiles are
// double-buffered in LDS via Tensor Data Mover DMA issued by wave 0
// (8x reduction of L2 traffic vs per-wave fetches).
// ---------------------------------------------------------------------------
__global__ __launch_bounds__(256) void attn_kernel(
    const _Float16* __restrict__ Qt, const _Float16* __restrict__ Kt,
    const _Float16* __restrict__ V, float* __restrict__ Sa) {
  __shared__ __align__(256) _Float16 Ks[2][32 * KROW];
  __shared__ __align__(256) _Float16 Vs[2][64 * VROW];
  const int wave = threadIdx.x >> 5;
  const int lane = threadIdx.x & 31;
  const int m    = lane & 15;
  const int h    = lane >> 4;
  const int b    = blockIdx.x >> 5;                       // 32 blocks / batch
  const int i0   = (((blockIdx.x & 31) << 3) + wave) << 4;
  const _Float16* Qb = Qt + (size_t)b * N_PIX * D_CH;
  const _Float16* Kb = Kt + (size_t)b * N_PIX * D_CH;
  const _Float16* Vb = V + (size_t)b * N_PIX * D_CH;
  const v16h qf0 = cat8(*(const v8h*)(Qb + (i0 + m) * 64 + h * 8), *(const v8h*)(Qb + (i0 + m) * 64 + 16 + h * 8));
  const v16h qf1 = cat8(*(const v8h*)(Qb + (i0 + m) * 64 + 32 + h * 8), *(const v8h*)(Qb + (i0 + m) * 64 + 48 + h * 8));
  v8f acc0 = {}, acc1 = {}, acc2 = {}, acc3 = {};
  float mrun = -__builtin_inff();
  float lrun = 0.0f;
  const float LOG2E = 1.44269504088896340736f;

  // preload j-block 0 into buffer 0
  if (wave == 0) {
    tdm_load_tile_2d((unsigned)(uintptr_t)&Ks[0][0], Kb, 64, N_PIX, 64, 32, 64,
                     4, 3);  // rows of 32 dwords, pad 4 dwords
    tdm_load_tile_2d((unsigned)(uintptr_t)&Vs[0][0], Vb, N_PIX, 64, 32, 64,
                     N_PIX, 3, 3);  // rows of 16 dwords, pad 4 dwords
    __builtin_amdgcn_s_wait_tensorcnt(0);
  }
  __syncthreads();

  for (int j0 = 0; j0 < N_PIX; j0 += 32) {
    const int cur = (j0 >> 5) & 1;
    if (wave == 0 && (j0 + 32) < N_PIX) {
      tdm_load_tile_2d((unsigned)(uintptr_t)&Ks[1 - cur][0],
                       Kb + (size_t)(j0 + 32) * 64, 64, N_PIX, 64, 32, 64, 4,
                       3);
      tdm_load_tile_2d((unsigned)(uintptr_t)&Vs[1 - cur][0], Vb + (j0 + 32),
                       N_PIX, 64, 32, 64, N_PIX, 3, 3);
    }
    // S^T fragments: lane holds column i=m; VGPR r holds row j=j0+r+8h (+16)
    const _Float16* k0 = &Ks[cur][m * KROW + h * 8];
    const _Float16* k1 = &Ks[cur][(16 + m) * KROW + h * 8];
    v16h a00 = cat8(*(const v8h*)(k0), *(const v8h*)(k0 + 16));
    v16h a01 = cat8(*(const v8h*)(k0 + 32), *(const v8h*)(k0 + 48));
    v16h a10 = cat8(*(const v8h*)(k1), *(const v8h*)(k1 + 16));
    v16h a11 = cat8(*(const v8h*)(k1 + 32), *(const v8h*)(k1 + 48));
    v8f slo = {}, shi = {};
    slo = wmma16(a00, qf0, slo);
    slo = wmma16(a01, qf1, slo);
    shi = wmma16(a10, qf0, shi);
    shi = wmma16(a11, qf1, shi);
    // online softmax over j for this lane's query i=m
    float mx = slo[0];
#pragma unroll
    for (int r = 1; r < 8; ++r) mx = fmaxf(mx, slo[r]);
#pragma unroll
    for (int r = 0; r < 8; ++r) mx = fmaxf(mx, shi[r]);
    mx = fmaxf(mx, __shfl_xor(mx, 16, 32));
    float mnew = fmaxf(mrun, mx);
    float plo[8], phi[8];
    float ls = 0.0f;
#pragma unroll
    for (int r = 0; r < 8; ++r) {
      plo[r] = exp2f((slo[r] - mnew) * LOG2E);
      phi[r] = exp2f((shi[r] - mnew) * LOG2E);
      ls += plo[r] + phi[r];
    }
    ls += __shfl_xor(ls, 16, 32);
    float scale = exp2f((mrun - mnew) * LOG2E);
    lrun = lrun * scale + ls;
    mrun = mnew;
    // broadcast per-row rescale to O-accumulator layout (row i = r + 8h)
    float sc8[8];
#pragma unroll
    for (int r = 0; r < 8; ++r) sc8[r] = __shfl(scale, r + 8 * h, 32);
#pragma unroll
    for (int r = 0; r < 8; ++r) {
      acc0[r] *= sc8[r]; acc1[r] *= sc8[r];
      acc2[r] *= sc8[r]; acc3[r] *= sc8[r];
    }
    // pack P (f32 C-frag of S^T) into f16 A-frag of P@V — pure in-register
    v16h af;
#pragma unroll
    for (int e = 0; e < 8; ++e) {
      af[e]     = (_Float16)plo[e];
      af[8 + e] = (_Float16)phi[e];
    }
    const _Float16* vp = &Vs[cur][m * VROW + h * 8];
    acc0 = wmma16(af, cat8(*(const v8h*)(vp), *(const v8h*)(vp + 16)), acc0);
    acc1 = wmma16(af, cat8(*(const v8h*)(vp + 16 * VROW), *(const v8h*)(vp + 16 * VROW + 16)), acc1);
    acc2 = wmma16(af, cat8(*(const v8h*)(vp + 32 * VROW), *(const v8h*)(vp + 32 * VROW + 16)), acc2);
    acc3 = wmma16(af, cat8(*(const v8h*)(vp + 48 * VROW), *(const v8h*)(vp + 48 * VROW + 16)), acc3);
    // next buffer must be resident before anyone proceeds
    if (wave == 0) __builtin_amdgcn_s_wait_tensorcnt(0);
    __syncthreads();
  }
  float* Sb = Sa + (size_t)b * N_PIX * D_CH;
#pragma unroll
  for (int r = 0; r < 8; ++r) {
    float li  = __shfl(lrun, r + 8 * h, 32);
    float inv = 1.0f / li;
    float* rowp = Sb + (size_t)(i0 + r + 8 * h) * 64 + m;
    rowp[0]  = acc0[r] * inv;
    rowp[16] = acc1[r] * inv;
    rowp[32] = acc2[r] * inv;
    rowp[48] = acc3[r] * inv;
  }
}

// ---------------------------------------------------------------------------
// Phase 3: out = gamma * (wsa @ sa + bsa) + x  via WMMA (M=c, K=d, N=n)
// ---------------------------------------------------------------------------
__global__ __launch_bounds__(256) void proj_kernel(
    const _Float16* __restrict__ wsah, const float* __restrict__ bsa,
    const float* __restrict__ gamma, const float* __restrict__ Sa,
    const float* __restrict__ x, float* __restrict__ out) {
  const int wid  = (blockIdx.x * 256 + threadIdx.x) >> 5;
  const int lane = threadIdx.x & 31;
  const int m    = lane & 15;
  const int h    = lane >> 4;
  const int b    = wid >> 12;
  const int c0   = ((wid >> 8) & 15) << 4;
  const int n0   = (wid & 255) << 4;
  const float* Sb = Sa + (size_t)b * N_PIX * D_CH;
  v8f acc = {};
#pragma unroll
  for (int s = 0; s < 2; ++s) {
    const _Float16* wp = wsah + (c0 + m) * 64 + s * 32 + h * 16;
    v16h a = cat8(*(const v8h*)wp, *(const v8h*)(wp + 8));
    const float* sp = Sb + (size_t)(n0 + m) * 64 + s * 32 + h * 16;
    v16h bf;
#pragma unroll
    for (int e = 0; e < 16; ++e) bf[e] = (_Float16)sp[e];
    acc = wmma16(a, bf, acc);
  }
  const float g = gamma[0];
#pragma unroll
  for (int r = 0; r < 8; ++r) {
    int cc = c0 + r + 8 * h;
    size_t off = ((size_t)(b * 256 + cc) << 12) + n0 + m;
    out[off] = g * (acc[r] + bsa[cc]) + x[off];
  }
}

// ---------------------------------------------------------------------------
extern "C" void kernel_launch(void* const* d_in, const int* in_sizes, int n_in,
                              void* d_out, int out_size, void* d_ws,
                              size_t ws_size, hipStream_t stream) {
  const float* x     = (const float*)d_in[0];
  const float* wq    = (const float*)d_in[1];
  const float* bq    = (const float*)d_in[2];
  const float* wk    = (const float*)d_in[3];
  const float* bk    = (const float*)d_in[4];
  const float* wv    = (const float*)d_in[5];
  const float* bv    = (const float*)d_in[6];
  const float* wsa   = (const float*)d_in[7];
  const float* bsa   = (const float*)d_in[8];
  const float* gamma = (const float*)d_in[9];

  char* ws = (char*)d_ws;
  _Float16* wqh  = (_Float16*)(ws);
  _Float16* wkh  = (_Float16*)(ws + 32768);
  _Float16* wvh  = (_Float16*)(ws + 65536);
  _Float16* wsah = (_Float16*)(ws + 98304);
  _Float16* Qt   = (_Float16*)(ws + 131072);
  _Float16* Kt   = (_Float16*)(ws + 131072 + 2097152);
  _Float16* V    = (_Float16*)(ws + 131072 + 2 * 2097152);
  float*    Sa   = (float*)(ws + 131072 + 3 * 2097152);
  float*    out  = (float*)d_out;

  prep_weights<<<64, 256, 0, stream>>>(wq, wk, wv, wsa, wqh, wkh, wvh, wsah);
  qkv_kernel<<<dim3(128, 4), 256, 0, stream>>>(x, wqh, wkh, wvh, bq, bk, bv,
                                               Qt, Kt, V);
  attn_kernel<<<128, 256, 0, stream>>>(Qt, Kt, V, Sa);
  proj_kernel<<<2048, 256, 0, stream>>>(wsah, bsa, gamma, Sa, x, out);
}
